// EdgeEnhancementBlock_90194313216173
// MI455X (gfx1250) — matrix operand order repaired and checked by reading, not compiled, not run on hardware
//
#include <hip/hip_runtime.h>

// Problem constants (from reference): x is (B, C, H, W) f32
#define Bn 16
#define Cn 192
#define Hn 128
#define Wn 128

typedef _Float16 v8h  __attribute__((ext_vector_type(8)));
typedef _Float16 v16h __attribute__((ext_vector_type(16)));
typedef float    v8f  __attribute__((ext_vector_type(8)));

// LDS tile stride (f16 elements) for t[pixel][channel]: 200*2B = 400B row
// pitch -> rows 16B-aligned (400 = 25*16); half-wave B-fragment reads touch
// each of the 64 LDS banks exactly twice (the minimum for 512B/wave).
#define TS 200

// ---------------------------------------------------------------------------
// Prekernel: convert merge weights f32 -> f16 into workspace (36864 elems,
// negligible; the 72KB f16 matrix then stays hot in L2/WGP$ for the GEMM).
// ---------------------------------------------------------------------------
__global__ __launch_bounds__(256)
void wcvt_kernel(const float* __restrict__ w, _Float16* __restrict__ wh, int n) {
    int i = blockIdx.x * blockDim.x + threadIdx.x;
    if (i < n) wh[i] = (_Float16)w[i];
}

// ---------------------------------------------------------------------------
// Fused kernel: one workgroup (8 waves) per (b, h) image row.
//   Phase 1: t = x + lap(x) = 5*x - up - down - left - right  -> LDS (f16)
//            left/right come from lane shuffles (only wave-edge lanes load).
//   Phase 2: per-wave WMMA GEMM over channels: y = W * t + bias.
//            B fragments preloaded once; A fragments streamed through an
//            explicit 3-deep rotating prefetch so loads stay ahead of WMMAs.
// ---------------------------------------------------------------------------
__global__ __launch_bounds__(256)
void edge_merge_kernel(const float* __restrict__ x,
                       const _Float16* __restrict__ wf16,
                       const float* __restrict__ bias,
                       float* __restrict__ out) {
    __shared__ _Float16 sT[Wn * TS];   // 128 pixels x 200 (192 ch + pad) = 50 KB

    const int bh   = blockIdx.x;
    const int b    = bh / Hn;
    const int h    = bh % Hn;
    const int tid  = threadIdx.x;
    const int lane = tid & 31;

    // ---- Phase 1: stage t for this row into LDS ([pixel][channel] f16) ----
    {
        const int wloc  = tid & (Wn - 1);   // pixel within row (coalesced)
        const int chalf = tid >> 7;         // 0/1: which of two channels/iter
        const float* xb = x + (size_t)b * Cn * Hn * Wn;
        #pragma unroll 4
        for (int it = 0; it < Cn / 2; ++it) {
            const int c = it * 2 + chalf;
            const float* xc = xb + (size_t)c * Hn * Wn + (size_t)h * Wn;
            float ctr = xc[wloc];
            float up = (h > 0)      ? xc[wloc - Wn] : 0.0f;
            float dn = (h < Hn - 1) ? xc[wloc + Wn] : 0.0f;
            // horizontal neighbors via wave32 shuffles; only edge lanes load
            float lf = __shfl_up(ctr, 1, 32);
            float rt = __shfl_down(ctr, 1, 32);
            if (lane == 0)  lf = (wloc > 0)      ? xc[wloc - 1] : 0.0f;
            if (lane == 31) rt = (wloc < Wn - 1) ? xc[wloc + 1] : 0.0f;
            float t = 5.0f * ctr - up - dn - lf - rt;   // x + 3x3 laplacian
            sT[wloc * TS + c] = (_Float16)t;
        }
    }
    __syncthreads();

    // ---- Phase 2: WMMA GEMM. Each wave: one 16-pixel N-tile, all 12 M-tiles.
    const int wave  = tid >> 5;
    const int lrow  = lane & 15;   // M (for A) / N (for B, C/D)
    const int lhalf = lane >> 4;   // which K/M half the lane holds
    const int wpix  = wave * 16 + lrow;  // this lane's pixel column (N)

    // Preload all 6 B fragments (32x16 each): lane half h holds
    // K = h*8..+7 and 16+h*8..+7 (same order as A).
    v16h bfrag[Cn / 32];
    #pragma unroll
    for (int k = 0; k < Cn / 32; ++k) {
        const _Float16* tp = &sT[wpix * TS + k * 32 + lhalf * 8];
        v8h blo = *(const v8h*)tp;
        v8h bhi = *(const v8h*)(tp + 16);
        bfrag[k] = __builtin_shufflevector(blo, bhi,
            0,1,2,3,4,5,6,7,8,9,10,11,12,13,14,15);
    }

    const _Float16* aw = wf16 + (size_t)lrow * Cn + lhalf * 8;
    float* ob = out + (size_t)b * Cn * Hn * Wn + (size_t)h * Wn + wpix;

    // A fragment (16x32) at flat index idx = m*6 + k: lane holds
    // K = half*8..+7 and 16+half*8..+7 of rows o = m*16 + lrow.
    auto loadA = [&](int idx) -> v16h {
        const int m = idx / 6, k = idx - m * 6;
        const _Float16* ap = aw + (size_t)(m * 16) * Cn + k * 32;
        v8h lo = *(const v8h*)ap;
        v8h hi = *(const v8h*)(ap + 16);
        return __builtin_shufflevector(lo, hi,
            0,1,2,3,4,5,6,7,8,9,10,11,12,13,14,15);
    };

    const v8f vzero = {};
    v16h a0 = loadA(0);
    v16h a1 = loadA(1);
    v16h a2 = loadA(2);
    v8f acc = vzero;

    // Flat 72-step pipeline: consume fragment idx, keep idx+1..idx+3 in
    // flight (6 outstanding b128 loads) so WMMAs never wait on loadcnt==0.
    #pragma unroll
    for (int idx = 0; idx < 72; ++idx) {
        const int k = idx % 6;
        v16h cur = a0;
        a0 = a1;
        a1 = a2;
        if (idx + 3 < 72) a2 = loadA(idx + 3);

        // D = A x B + C, f32 accumulate
        acc = __builtin_amdgcn_wmma_f32_16x16x32_f16(
            false, cur, false, bfrag[k], (short)0, acc, false, false);

        if (k == 5) {   // finished one 16x16 output tile -> bias + store
            const int m = idx / 6;
            // 8 consecutive output channels per lane half: vector bias load
            v8f bi = *(const v8f*)(bias + m * 16 + lhalf * 8);  // 32B aligned
            #pragma unroll
            for (int i = 0; i < 8; ++i) {
                const int o = m * 16 + lhalf * 8 + i;  // VGPR i -> M=i+8*half
                ob[(size_t)o * Hn * Wn] = acc[i] + bi[i];
            }
            acc = vzero;
        }
    }
}

extern "C" void kernel_launch(void* const* d_in, const int* in_sizes, int n_in,
                              void* d_out, int out_size, void* d_ws, size_t ws_size,
                              hipStream_t stream) {
    const float* x  = (const float*)d_in[0];   // (16,192,128,128) f32
    const float* mw = (const float*)d_in[1];   // (192,192,1,1) f32
    const float* mb = (const float*)d_in[2];   // (192,) f32
    float* out = (float*)d_out;                // (16,192,128,128) f32

    _Float16* wh = (_Float16*)d_ws;            // 192*192 f16 = 73 728 B
    const int nw = Cn * Cn;
    wcvt_kernel<<<(nw + 255) / 256, 256, 0, stream>>>(mw, wh, nw);

    edge_merge_kernel<<<Bn * Hn, 256, 0, stream>>>(x, wh, mb, out);
}
